// SocialAggregation_46076409152412
// MI455X (gfx1250) — matrix operand order repaired and checked by reading, not compiled, not run on hardware
//
#include <hip/hip_runtime.h>
#include <hip/hip_bf16.h>

#define D_DIM   128
#define H_DIM   256
#define K_NBR   32
#define TWO_D   256
#define A_STRIDE 264                       // halves per A row (256 + 8 pad) = 528 B, 16B aligned
#define WAVE_LDS_BYTES (K_NBR * A_STRIDE * 2)   // 16896 B per wave
#define WAVES_PER_BLOCK 4

typedef __attribute__((ext_vector_type(16))) __bf16 v16bf;
typedef __attribute__((ext_vector_type(8)))  __bf16 v8bf;
typedef __attribute__((ext_vector_type(4)))  __bf16 v4bf;
typedef __attribute__((ext_vector_type(8)))  float  v8f;
typedef __attribute__((ext_vector_type(4)))  float  v4f;

// ---------------------------------------------------------------------------
// Pre-pack W1 (fp32 [2D, H] row-major) into bf16 WMMA B-operand fragments.
// Fragment (nt, ks), lane L, half h:
//   value = W1[ ks*32 + (L>>4)*16 + h ][ nt*16 + (L&15) ]
// contiguous k order k = 16*(L>>4) + h, same as the A-tile loads. 16*8 frags * 32 lanes * 16
// halves = 65536 bf16 = 128 KB in d_ws (L2-resident, shared by all waves).
// ---------------------------------------------------------------------------
__global__ void sa_prep_w1(const float* __restrict__ W1, __bf16* __restrict__ wsB) {
  int idx = blockIdx.x * blockDim.x + threadIdx.x;   // 0..65535
  int h   = idx & 15;
  int L   = (idx >> 4) & 31;
  int ks  = (idx >> 9) & 7;
  int nt  = idx >> 12;
  int n   = nt * 16 + (L & 15);
  int kg  = ks * 32 + (L >> 4) * 16 + h;
  wsB[idx] = (__bf16)W1[kg * H_DIM + n];
}

// ---------------------------------------------------------------------------
// One wave32 per user. 4 waves / 128-thread block.
// ---------------------------------------------------------------------------
__global__ __launch_bounds__(128) void sa_main(
    const float* __restrict__ user_emb_pi,   // [N_USERS, D]
    const float* __restrict__ h_I_all,       // [N_USERS, D]
    const int*   __restrict__ user_idx,      // [B]
    const int*   __restrict__ neighbors,     // [B, K]
    const unsigned char* __restrict__ nmask, // [B, K] (bool)
    const __bf16* __restrict__ wsB,          // packed W1 fragments
    const float* __restrict__ b1,            // [H]
    const float* __restrict__ W2,            // [H]
    const float* __restrict__ b2,            // [1]
    const float* __restrict__ W_agg,         // [D, D] row-major
    const float* __restrict__ b_lin,         // [D]
    const float* __restrict__ b_agg,         // [D]
    float* __restrict__ out,                 // [B, D]
    int Bn)
{
  extern __shared__ char smem[];
  const int lane = threadIdx.x & 31;
  const int wave = threadIdx.x >> 5;
  const int b    = blockIdx.x * WAVES_PER_BLOCK + wave;
  const bool active = (b < Bn);

  __bf16* A = (__bf16*)(smem + wave * WAVE_LDS_BYTES);   // A tile [32][A_STRIDE] bf16

  // ---- Phase 1: gather neighbors + broadcast pi -> bf16 concat tile in LDS
  if (active) {
    const int uix = user_idx[b];
    v4f pv = *(const v4f*)(user_emb_pi + (size_t)uix * D_DIM + lane * 4);
    v4bf pb;
    #pragma unroll
    for (int i = 0; i < 4; ++i) pb[i] = (__bf16)pv[i];

    #pragma unroll 4
    for (int k = 0; k < K_NBR; ++k) {
      const int nb = neighbors[b * K_NBR + k];
      v4f hv = *(const v4f*)(h_I_all + (size_t)nb * D_DIM + lane * 4);
      v4bf hb;
      #pragma unroll
      for (int i = 0; i < 4; ++i) hb[i] = (__bf16)hv[i];
      *(v4bf*)(A + k * A_STRIDE + lane * 4)         = hb;  // h_O part (cols 0..127)
      *(v4bf*)(A + k * A_STRIDE + D_DIM + lane * 4) = pb;  // pi part  (cols 128..255)
    }
  }
  __syncthreads();

  // ---- Phase 2: hid = relu(A @ W1 + b1) fused with beta* = hid . W2, via WMMA
  float pb0[8], pb1[8];
  #pragma unroll
  for (int r = 0; r < 8; ++r) { pb0[r] = 0.f; pb1[r] = 0.f; }

  const int half = lane >> 4;    // 0: K 0-7/16-23 side, 1: K 8-15/24-31 side
  const int lq   = lane & 15;

  if (active) {
    for (int nt = 0; nt < 16; ++nt) {
      v8f c0 = {}; v8f c1 = {};
      #pragma unroll
      for (int ks = 0; ks < 8; ++ks) {
        const int col = ks * 32 + half * 16;
        // A operand, M-tile 0 (rows 0..15): lane holds row lq
        v8bf a0lo = *(const v8bf*)(A + lq * A_STRIDE + col);
        v8bf a0hi = *(const v8bf*)(A + lq * A_STRIDE + col + 8);
        v16bf a0  = __builtin_shufflevector(a0lo, a0hi,
                      0,1,2,3,4,5,6,7,8,9,10,11,12,13,14,15);
        // A operand, M-tile 1 (rows 16..31)
        v8bf a1lo = *(const v8bf*)(A + (16 + lq) * A_STRIDE + col);
        v8bf a1hi = *(const v8bf*)(A + (16 + lq) * A_STRIDE + col + 8);
        v16bf a1  = __builtin_shufflevector(a1lo, a1hi,
                      0,1,2,3,4,5,6,7,8,9,10,11,12,13,14,15);
        // B operand fragment: contiguous 32 B per lane
        v16bf bf = *(const v16bf*)(wsB + ((size_t)(nt * 8 + ks) * 32 + lane) * 16);

        c0 = __builtin_amdgcn_wmma_f32_16x16x32_bf16(
                 false, a0, false, bf, (short)0, c0, false, false);
        c1 = __builtin_amdgcn_wmma_f32_16x16x32_bf16(
                 false, a1, false, bf, (short)0, c1, false, false);
      }
      // fuse: relu(+b1) then dot with W2 for this N-column
      const int n = nt * 16 + lq;
      const float b1v = b1[n];
      const float w2v = W2[n];
      #pragma unroll
      for (int r = 0; r < 8; ++r) {
        pb0[r] += fmaxf(c0[r] + b1v, 0.f) * w2v;   // row k = half*8 + r
        pb1[r] += fmaxf(c1[r] + b1v, 0.f) * w2v;   // row k = 16 + half*8 + r
      }
    }
  }

  // butterfly-reduce over the 16 lanes of each half (sums the N dimension)
  #pragma unroll
  for (int m = 1; m < 16; m <<= 1) {
    #pragma unroll
    for (int r = 0; r < 8; ++r) {
      pb0[r] += __shfl_xor(pb0[r], m, 32);
      pb1[r] += __shfl_xor(pb1[r], m, 32);
    }
  }

  // ---- softmax over K=32 (distributed: this lane holds k in
  //      {8*half..8*half+7} U {16+8*half..16+8*half+7}; mask-aware)
  float* beta_buf = (float*)A;        // A tile is dead; reuse its LDS
  float* sw_buf   = beta_buf + 64;
  if (active) {
    const float b2v = b2[0];
    float v0[8], v1[8];
    unsigned char m0[8], m1[8];
    float mx = -__builtin_inff();
    #pragma unroll
    for (int r = 0; r < 8; ++r) {
      const int k0 = half * 8 + r;
      const int k1 = 16 + half * 8 + r;
      m0[r] = nmask[b * K_NBR + k0];
      m1[r] = nmask[b * K_NBR + k1];
      v0[r] = m0[r] ? (pb0[r] + b2v) : -__builtin_inff();
      v1[r] = m1[r] ? (pb1[r] + b2v) : -__builtin_inff();
      mx = fmaxf(mx, fmaxf(v0[r], v1[r]));
    }
    mx = fmaxf(mx, __shfl_xor(mx, 16, 32));
    float e0[8], e1[8], s = 0.f;
    #pragma unroll
    for (int r = 0; r < 8; ++r) {
      e0[r] = m0[r] ? __expf(v0[r] - mx) : 0.f;
      e1[r] = m1[r] ? __expf(v1[r] - mx) : 0.f;
      s += e0[r] + e1[r];
    }
    s += __shfl_xor(s, 16, 32);
    const float inv = (s > 0.f) ? (1.f / s) : 0.f;   // no-neighbor user -> zeros
    if (lq == 0) {
      #pragma unroll
      for (int r = 0; r < 8; ++r) {
        beta_buf[half * 8 + r]      = e0[r] * inv;
        beta_buf[16 + half * 8 + r] = e1[r] * inv;
      }
    }
  }
  __syncthreads();

  // ---- Phase 3: sum_weighted[d] = sum_k beta[k] * h_O[k][d]   (fp32 from L2)
  if (active) {
    v4f sw = {};
    #pragma unroll 4
    for (int k = 0; k < K_NBR; ++k) {
      const float w = beta_buf[k];
      const int nb = neighbors[b * K_NBR + k];
      v4f hv = *(const v4f*)(h_I_all + (size_t)nb * D_DIM + lane * 4);
      sw += w * hv;
    }
    *(v4f*)(sw_buf + lane * 4) = sw;
  }
  __syncthreads();

  // ---- Phase 4: out = relu(sum_weighted @ W_agg^T + b_lin + b_agg)
  if (active) {
    #pragma unroll
    for (int i = 0; i < 4; ++i) {
      const int d = lane * 4 + i;
      float acc = b_lin[d] + b_agg[d];
      #pragma unroll 8
      for (int j = 0; j < D_DIM; j += 4) {
        v4f wr = *(const v4f*)(W_agg + (size_t)d * D_DIM + j);
        v4f sv = *(const v4f*)(sw_buf + j);
        acc += wr[0] * sv[0] + wr[1] * sv[1] + wr[2] * sv[2] + wr[3] * sv[3];
      }
      out[(size_t)b * D_DIM + d] = fmaxf(acc, 0.f);
    }
  }
}

// ---------------------------------------------------------------------------
extern "C" void kernel_launch(void* const* d_in, const int* in_sizes, int n_in,
                              void* d_out, int out_size, void* d_ws, size_t ws_size,
                              hipStream_t stream) {
  const float*         user_emb_pi = (const float*)d_in[0];
  const float*         h_I_all     = (const float*)d_in[1];
  const int*           user_idx    = (const int*)d_in[2];
  const int*           neighbors   = (const int*)d_in[3];
  const unsigned char* nmask       = (const unsigned char*)d_in[4];
  const float*         W1          = (const float*)d_in[5];
  const float*         b1          = (const float*)d_in[6];
  const float*         W2          = (const float*)d_in[7];
  const float*         b2          = (const float*)d_in[8];
  const float*         W_agg       = (const float*)d_in[9];
  const float*         b_lin       = (const float*)d_in[10];
  const float*         b_agg       = (const float*)d_in[11];
  float* out = (float*)d_out;
  const int B = in_sizes[2];                 // user_idx element count

  __bf16* wsB = (__bf16*)d_ws;               // 65536 bf16 = 128 KB

  // pack W1 into WMMA B fragments (deterministic, every launch)
  sa_prep_w1<<<256, 256, 0, stream>>>(W1, wsB);

  const int blocks = (B + WAVES_PER_BLOCK - 1) / WAVES_PER_BLOCK;
  const size_t shmem = (size_t)WAVES_PER_BLOCK * WAVE_LDS_BYTES;   // 67584 B
  sa_main<<<blocks, WAVES_PER_BLOCK * 32, shmem, stream>>>(
      user_emb_pi, h_I_all, user_idx, neighbors, nmask, wsB,
      b1, W2, b2, W_agg, b_lin, b_agg, out, B);
}
